// DifferentialAttention_70858370449567
// MI455X (gfx1250) — matrix operand order repaired and checked by reading, not compiled, not run on hardware
//
#include <hip/hip_runtime.h>
#include <hip/hip_bf16.h>

// ---------------------------------------------------------------------------
// Differential attention for MI455X (gfx1250), wave32 + WMMA f16 (16x16x32).
//
// GEMMs stage block tiles into DOUBLE-BUFFERED LDS with gfx1250 async copies
// (global_load_async_to_lds_b128 / s_wait_asynccnt): the 16 KB tile fetch for
// step kt+1 is in flight while step kt's 8 WMMAs execute. 8-wave tile reuse
// gives 64 FLOP per global byte.
//
// Attention is computed fully TRANSPOSED (S^T = K@Q^T, O^T = V^T@P^T) so the
// online-softmax state per query is a per-lane scalar; cross-lane traffic is
// 4 shfl_xor(16) per 32-key step.
// ---------------------------------------------------------------------------

typedef _Float16 h16 __attribute__((ext_vector_type(16)));
typedef _Float16 h8  __attribute__((ext_vector_type(8)));
typedef float    f8  __attribute__((ext_vector_type(8)));

#define S_SEQ  2048
#define DMODEL 2048
#define NHEAD  16
#define DHEAD  64
#define LAMBDA_INIT 0.8f

// gfx1250 async global->LDS staging (ASYNCcnt-tracked). Fallback: sync copy.
#define ASYNC_STAGE 1

static __device__ __forceinline__ f8 wmma_f16(h16 a, h16 b, f8 c) {
  // v_wmma_f32_16x16x32_f16  (neg_a, A, neg_b, B, c_mod, C, reuse_a, reuse_b)
  return __builtin_amdgcn_wmma_f32_16x16x32_f16(false, a, false, b, (short)0, c,
                                                false, false);
}

#if ASYNC_STAGE
// One 16-byte async copy: LDS[lds_off] = MEM[gbase + goff]  (GVS addressing)
static __device__ __forceinline__ void async_b128(unsigned lds_off, unsigned goff,
                                                  const void* gbase) {
  asm volatile("global_load_async_to_lds_b128 %0, %1, %2"
               :: "v"(lds_off), "v"(goff), "s"(gbase) : "memory");
}
#endif

// ---------------------------------------------------------------------------
// lambda scalar
// ---------------------------------------------------------------------------
__global__ void lam_kernel(const float* __restrict__ lq1, const float* __restrict__ lk1,
                           const float* __restrict__ lq2, const float* __restrict__ lk2,
                           float* __restrict__ lam) {
  if (threadIdx.x == 0) {
    float s1 = 0.f, s2 = 0.f;
    for (int i = 0; i < DHEAD; ++i) { s1 += lq1[i] * lk1[i]; s2 += lq2[i] * lk2[i]; }
    lam[0] = __expf(s1) - __expf(s2) + LAMBDA_INIT;
  }
}

// ---------------------------------------------------------------------------
// Pack X (f32 row-major [M][K]) into A-fragment tiles, K-MAJOR tile order:
//   flat tile = kt*MT + mt   (so a block's 8 A tiles at one kt are contiguous)
// A layout (16-bit A 16x32, wave32): lane L -> row = L&15, g = L>>4,
//   element e -> k = (e>>3)*16 + g*8 + (e&7)
// ---------------------------------------------------------------------------
__global__ void pack_A_f32(const float* __restrict__ X, _Float16* __restrict__ Ap,
                           int Mt, int Kt, int ldx) {
  int tile = blockIdx.x * (blockDim.x >> 5) + (threadIdx.x >> 5);
  if (tile >= Mt * Kt) return;
  int kt = tile / Mt, mt = tile % Mt;          // K-major
  int lane = threadIdx.x & 31, g = lane >> 4, ln = lane & 15;
  const float* src = X + (size_t)(mt * 16 + ln) * ldx + kt * 32 + g * 8;
  h16 fr;
  #pragma unroll
  for (int i = 0; i < 8; ++i) { fr[i] = (_Float16)src[i]; fr[i + 8] = (_Float16)src[i + 16]; }
  *(h16*)(Ap + (size_t)tile * 512 + lane * 16) = fr;
}

// ---------------------------------------------------------------------------
// Pack W (f32 row-major [K][N]) into B-fragment tiles (flat = kt*NT + nt).
// B layout (16-bit B 32x16, wave32): lane L -> col = L&15, g = L>>4,
//   element e -> k = (e>>3)*16 + g*8 + (e&7)
// ---------------------------------------------------------------------------
__global__ void pack_B_f32(const float* __restrict__ W, _Float16* __restrict__ Bp,
                           int Kt, int Nt, int ldw) {
  int tile = blockIdx.x * (blockDim.x >> 5) + (threadIdx.x >> 5);
  if (tile >= Kt * Nt) return;
  int kt = tile / Nt, nt = tile % Nt;
  int lane = threadIdx.x & 31, g = lane >> 4, ln = lane & 15;
  const float* col = W + (size_t)(kt * 32 + g * 8) * ldw + nt * 16 + ln;
  h16 fr;
  #pragma unroll
  for (int e = 0; e < 16; ++e) fr[e] = (_Float16)col[(size_t)(((e >> 3) << 4) + (e & 7)) * ldw];
  *(h16*)(Bp + (size_t)tile * 512 + lane * 16) = fr;
}

// ---------------------------------------------------------------------------
// WMMA GEMM with double-buffered async LDS staging.
// Block: 256 threads = 8 waves, 4(M)x2(N); wave tile 32x64 (2x4 WMMA tiles).
// Block tile: 128(M) x 128(N). Per 32-K step: 16 KB staged once into LDS
// (each thread: 4 async b128), shared by all 8 waves; the copy for step kt+1
// overlaps the WMMAs of step kt.
// ---------------------------------------------------------------------------
template <typename OutT>
__global__ void __launch_bounds__(256)
gemm_wmma(const _Float16* __restrict__ Ap, const _Float16* __restrict__ Bp,
          OutT* __restrict__ C, int M, int N, int K) {
  __shared__ __align__(128) _Float16 sA[2][8 * 512];   // 2 x 8 KB
  __shared__ __align__(128) _Float16 sB[2][8 * 512];   // 2 x 8 KB

  const int KT = K >> 5, MT = M >> 4, NT = N >> 4;
  const int td = threadIdx.x;
  const int w = td >> 5, lane = td & 31, g = lane >> 4, ln = lane & 15;
  const int wm = w >> 1, wn = w & 1;
  const int mtB = blockIdx.y * 8;   // first M-tile of this block
  const int ntB = blockIdx.x * 8;   // first N-tile of this block

#if ASYNC_STAGE
  auto stage = [&](int buf, int kt) {
    unsigned aoff = (unsigned)((kt * MT + mtB) * 1024 + td * 32);
    unsigned boff = (unsigned)((kt * NT + ntB) * 1024 + td * 32);
    unsigned la = (unsigned)(uintptr_t)(&sA[buf][0]) + (unsigned)td * 32;
    unsigned lb = (unsigned)(uintptr_t)(&sB[buf][0]) + (unsigned)td * 32;
    async_b128(la,      aoff,      Ap);
    async_b128(la + 16, aoff + 16, Ap);
    async_b128(lb,      boff,      Bp);
    async_b128(lb + 16, boff + 16, Bp);
  };
#endif

  f8 acc[2][4] = {};

#if ASYNC_STAGE
  stage(0, 0);  // prologue: copies for kt=0 in flight
#endif

  for (int kt = 0; kt < KT; ++kt) {
    const int cur = kt & 1;
#if ASYNC_STAGE
    if (kt + 1 < KT) {
      stage(cur ^ 1, kt + 1);                          // next tile in flight
      asm volatile("s_wait_asynccnt 0x4" ::: "memory"); // older 4 (cur buf) done
    } else {
      asm volatile("s_wait_asynccnt 0x0" ::: "memory");
    }
#else
    {
      unsigned aoff = (unsigned)((kt * MT + mtB) * 1024 + td * 32);
      unsigned boff = (unsigned)((kt * NT + ntB) * 1024 + td * 32);
      *(h16*)(&sA[cur][0] + td * 16) = *(const h16*)((const char*)Ap + aoff);
      *(h16*)(&sB[cur][0] + td * 16) = *(const h16*)((const char*)Bp + boff);
    }
#endif
    __syncthreads();

    // ---- consume: 2 A frags + 4 B frags from LDS, 8 WMMAs ----
    h16 a0 = *(const h16*)(&sA[cur][0] + (size_t)(wm * 2 + 0) * 512 + lane * 16);
    h16 a1 = *(const h16*)(&sA[cur][0] + (size_t)(wm * 2 + 1) * 512 + lane * 16);
    h16 b[4];
    #pragma unroll
    for (int j = 0; j < 4; ++j)
      b[j] = *(const h16*)(&sB[cur][0] + (size_t)(wn * 4 + j) * 512 + lane * 16);

    #pragma unroll
    for (int j = 0; j < 4; ++j) {
      acc[0][j] = wmma_f16(a0, b[j], acc[0][j]);
      acc[1][j] = wmma_f16(a1, b[j], acc[1][j]);
    }

    // all LDS reads drained before this buffer is overwritten (kt+2's stage)
    asm volatile("s_wait_dscnt 0x0" ::: "memory");
    __syncthreads();
  }

  // C/D layout: lane -> col = ln, vgpr v -> row = v + 8*g
  #pragma unroll
  for (int i = 0; i < 2; ++i)
    #pragma unroll
    for (int j = 0; j < 4; ++j) {
      int colb = (ntB + wn * 4 + j) * 16 + ln;
      #pragma unroll
      for (int v = 0; v < 8; ++v) {
        int row = (mtB + wm * 2 + i) * 16 + v + g * 8;
        C[(size_t)row * N + colb] = (OutT)acc[i][j][v];
      }
    }
}

// ---------------------------------------------------------------------------
// Per-head K pack as A-fragments for S^T = K @ Q^T.
// A tile: 16 keys (rows) x 32 dh. Flat: ((h*2+comp)*128 + keytile)*2 + kt2
// ---------------------------------------------------------------------------
__global__ void pack_Khead(const _Float16* __restrict__ Kb, _Float16* __restrict__ Kp) {
  int tile = blockIdx.x * (blockDim.x >> 5) + (threadIdx.x >> 5);
  if (tile >= NHEAD * 2 * 128 * 2) return;
  int kt2     = tile & 1;
  int keytile = (tile >> 1) & 127;
  int comp    = (tile >> 8) & 1;
  int h       = tile >> 9;
  int lane = threadIdx.x & 31, g = lane >> 4, ln = lane & 15;
  const _Float16* src = Kb + (size_t)(keytile * 16 + ln) * DMODEL +
                        h * 128 + comp * 64 + kt2 * 32 + g * 16;
  h8 lo = *(const h8*)src;
  h8 hi = *(const h8*)(src + 8);
  h16 fr;
  #pragma unroll
  for (int i = 0; i < 8; ++i) { fr[i] = lo[i]; fr[i + 8] = hi[i]; }
  *(h16*)(Kp + (size_t)tile * 512 + lane * 16) = fr;
}

// ---------------------------------------------------------------------------
// Per-head V^T pack as A-fragments for O^T = V^T @ P^T.
// A tile: 16 channels (rows) x 32 keys. Flat: (h*64 + kt32)*8 + ct
// ---------------------------------------------------------------------------
__global__ void pack_Vhead(const _Float16* __restrict__ Vb, _Float16* __restrict__ Vp) {
  int tile = blockIdx.x * (blockDim.x >> 5) + (threadIdx.x >> 5);
  if (tile >= NHEAD * 64 * 8) return;
  int ct   = tile & 7;
  int kt32 = (tile >> 3) & 63;
  int h    = tile >> 9;
  int lane = threadIdx.x & 31, g = lane >> 4, ln = lane & 15;
  int ch = h * 128 + ct * 16 + ln;
  h16 fr;
  #pragma unroll
  for (int e = 0; e < 16; ++e) {
    int key = kt32 * 32 + g * 16 + e;  // A-layout k map (contiguous)
    fr[e] = Vb[(size_t)key * DMODEL + ch];
  }
  *(h16*)(Vp + (size_t)tile * 512 + lane * 16) = fr;
}

// ---------------------------------------------------------------------------
// Dual flash attention (transposed) + lambda combine + GroupNorm.
// One wave per (head, 16-query block). Queries map to lanes; per-lane scalar
// softmax state; probs bounce via wave-private LDS into B-fragment layout.
// Emits A-fragment-packed f16 in K-MAJOR tile order: (h*4+ktl)*128 + qb.
// ---------------------------------------------------------------------------
__global__ void __launch_bounds__(32)
diff_attn_kernel(const _Float16* __restrict__ Qb, const _Float16* __restrict__ Kp,
                 const _Float16* __restrict__ Vp, const float* __restrict__ lamp,
                 const float* __restrict__ gnw, const float* __restrict__ gnb,
                 _Float16* __restrict__ ApAtt) {
  __shared__ __align__(32) _Float16 lp1[16 * 32];
  __shared__ __align__(32) _Float16 lp2[16 * 32];
  __shared__ __align__(32) _Float16 lout[16 * 128];

  const int h = blockIdx.y, qb = blockIdx.x;
  const int lane = threadIdx.x & 31, g = lane >> 4, ln = lane & 15;
  const float lam = lamp[0];

  // Q^T B-fragments [comp][dh-half], scale 1/sqrt(64)=2^-3 folded in (exact).
  h16 qf[2][2];
  #pragma unroll
  for (int comp = 0; comp < 2; ++comp)
    #pragma unroll
    for (int kt2 = 0; kt2 < 2; ++kt2) {
      h16 raw = *(const h16*)(Qb + (size_t)(qb * 16 + ln) * DMODEL +
                              h * 128 + comp * 64 + kt2 * 32 + g * 16);
      #pragma unroll
      for (int i = 0; i < 16; ++i) qf[comp][kt2][i] = (_Float16)((float)raw[i] * 0.125f);
    }

  f8 o1[8] = {}, o2[8] = {};
  float rm1 = -1e30f, rl1 = 0.f, rm2 = -1e30f, rl2 = 0.f;

  const _Float16* KpH = Kp + (size_t)h * 2 * 128 * 2 * 512;

  for (int kt32 = 0; kt32 < 64; ++kt32) {
    // ---- S^T tiles: rows = keys (v + 8g + 16t), cols = queries (ln) ----
    f8 s1[2], s2[2];
    #pragma unroll
    for (int t = 0; t < 2; ++t) {
      int keytile = kt32 * 2 + t;
      const _Float16* ka1 = KpH + ((size_t)(0 * 128 + keytile) * 2) * 512 + lane * 16;
      const _Float16* ka2 = KpH + ((size_t)(1 * 128 + keytile) * 2) * 512 + lane * 16;
      f8 z = {};
      s1[t] = wmma_f16(*(const h16*)(ka1 + 512), qf[0][1],
               wmma_f16(*(const h16*)ka1, qf[0][0], z));
      s2[t] = wmma_f16(*(const h16*)(ka2 + 512), qf[1][1],
               wmma_f16(*(const h16*)ka2, qf[1][0], z));
    }

    // ---- per-query (per-lane) online max ----
    float m1 = -1e30f, m2 = -1e30f;
    #pragma unroll
    for (int t = 0; t < 2; ++t)
      #pragma unroll
      for (int v = 0; v < 8; ++v) { m1 = fmaxf(m1, s1[t][v]); m2 = fmaxf(m2, s2[t][v]); }
    m1 = fmaxf(m1, __shfl_xor(m1, 16, 32));
    m2 = fmaxf(m2, __shfl_xor(m2, 16, 32));
    float nm1 = fmaxf(rm1, m1), nm2 = fmaxf(rm2, m2);
    float cf1 = __expf(rm1 - nm1), cf2 = __expf(rm2 - nm2);
    rm1 = nm1;  rm2 = nm2;

    // ---- exp + write P^T to LDS as [query][key] (contiguous h8 stores) ----
    __syncthreads();
    float ps1 = 0.f, ps2 = 0.f;
    #pragma unroll
    for (int t = 0; t < 2; ++t) {
      h8 p1v, p2v;
      #pragma unroll
      for (int v = 0; v < 8; ++v) {
        float p1 = __expf(s1[t][v] - nm1);
        float p2 = __expf(s2[t][v] - nm2);
        ps1 += p1;  ps2 += p2;
        p1v[v] = (_Float16)p1;  p2v[v] = (_Float16)p2;
      }
      *(h8*)&lp1[ln * 32 + t * 16 + 8 * g] = p1v;
      *(h8*)&lp2[ln * 32 + t * 16 + 8 * g] = p2v;
    }
    __syncthreads();
    ps1 += __shfl_xor(ps1, 16, 32);
    ps2 += __shfl_xor(ps2, 16, 32);
    rl1 = rl1 * cf1 + ps1;
    rl2 = rl2 * cf2 + ps2;

    // ---- P^T B-fragments: k = g*16+e over keys, col = query ln ----
    h16 b1 = *(const h16*)&lp1[ln * 32 + g * 16];
    h16 b2 = *(const h16*)&lp2[ln * 32 + g * 16];

    // ---- O^T accumulation: per-lane scalar rescale, then WMMA ----
    const _Float16* vbase = Vp + (size_t)(h * 64 + kt32) * 8 * 512 + lane * 16;
    #pragma unroll
    for (int ct = 0; ct < 8; ++ct) {
      #pragma unroll
      for (int v = 0; v < 8; ++v) { o1[ct][v] *= cf1; o2[ct][v] *= cf2; }
      h16 va = *(const h16*)(vbase + (size_t)ct * 512);
      o1[ct] = wmma_f16(va, b1, o1[ct]);
      o2[ct] = wmma_f16(va, b2, o2[ct]);
    }
  }

  // ---- epilogue: normalize, lambda-combine, GroupNorm per query ----
  float inv1 = 1.f / rl1, inv2 = 1.f / rl2;
  float outv[8][8];
  float s = 0.f, sq = 0.f;
  #pragma unroll
  for (int ct = 0; ct < 8; ++ct)
    #pragma unroll
    for (int v = 0; v < 8; ++v) {
      float y = o1[ct][v] * inv1 - lam * (o2[ct][v] * inv2);
      outv[ct][v] = y;  s += y;  sq += y * y;
    }
  s  += __shfl_xor(s, 16, 32);
  sq += __shfl_xor(sq, 16, 32);
  float mu  = s * (1.f / 128.f);
  float var = sq * (1.f / 128.f) - mu * mu;
  float rs  = rsqrtf(var + 1e-5f);

  __syncthreads();
  #pragma unroll
  for (int ct = 0; ct < 8; ++ct) {
    h8 yv;
    #pragma unroll
    for (int v = 0; v < 8; ++v) {
      int ch = ct * 16 + 8 * g + v;  // channel of this element
      float y = ((outv[ct][v] - mu) * rs * gnw[h * 128 + ch] + gnb[h * 128 + ch]) *
                (1.0f - LAMBDA_INIT);
      yv[v] = (_Float16)y;
    }
    *(h8*)&lout[ln * 128 + ct * 16 + 8 * g] = yv;
  }
  __syncthreads();

  // ---- emit A-fragment-packed tiles, K-major: flat = (h*4+ktl)*128 + qb ----
  #pragma unroll
  for (int ktl = 0; ktl < 4; ++ktl) {
    h8 lo = *(const h8*)&lout[ln * 128 + ktl * 32 + g * 8];
    h8 hi = *(const h8*)&lout[ln * 128 + ktl * 32 + g * 8 + 16];
    h16 fr;
    #pragma unroll
    for (int i = 0; i < 8; ++i) { fr[i] = lo[i]; fr[i + 8] = hi[i]; }
    *(h16*)(ApAtt + ((size_t)(h * 4 + ktl) * 128 + qb) * 512 + lane * 16) = fr;
  }
}

// ---------------------------------------------------------------------------
// Host-side launch
// ---------------------------------------------------------------------------
extern "C" void kernel_launch(void* const* d_in, const int* in_sizes, int n_in,
                              void* d_out, int out_size, void* d_ws, size_t ws_size,
                              hipStream_t stream) {
  (void)in_sizes; (void)n_in; (void)out_size; (void)ws_size;
  const float* x   = (const float*)d_in[0];
  const float* Wq  = (const float*)d_in[1];
  const float* Wk  = (const float*)d_in[2];
  const float* Wv  = (const float*)d_in[3];
  const float* Wo  = (const float*)d_in[4];
  const float* lq1 = (const float*)d_in[5];
  const float* lk1 = (const float*)d_in[6];
  const float* lq2 = (const float*)d_in[7];
  const float* lk2 = (const float*)d_in[8];
  const float* gnw = (const float*)d_in[9];
  const float* gnb = (const float*)d_in[10];
  float* out = (float*)d_out;

  // Workspace carve-out (~92 MB total).
  char* ws = (char*)d_ws;
  size_t off = 0;
  auto alloc = [&](size_t bytes) -> char* {
    char* p = ws + off;
    off = (off + bytes + 255) & ~(size_t)255;
    return p;
  };
  const size_t PACK = (size_t)128 * 64 * 512 * sizeof(_Float16);  // 8 MB
  const size_t ROWM = (size_t)S_SEQ * DMODEL * sizeof(_Float16);  // 8 MB

  float*    lam   = (float*)alloc(256);
  _Float16* ApX   = (_Float16*)alloc(PACK);
  _Float16* BpQ   = (_Float16*)alloc(PACK);
  _Float16* BpK   = (_Float16*)alloc(PACK);
  _Float16* BpV   = (_Float16*)alloc(PACK);
  _Float16* BpO   = (_Float16*)alloc(PACK);
  _Float16* Qb    = (_Float16*)alloc(ROWM);
  _Float16* Kb    = (_Float16*)alloc(ROWM);
  _Float16* Vb    = (_Float16*)alloc(ROWM);
  _Float16* Kp    = (_Float16*)alloc(PACK);
  _Float16* Vp    = (_Float16*)alloc(PACK);
  _Float16* ApAtt = (_Float16*)alloc(PACK);

  lam_kernel<<<1, 1, 0, stream>>>(lq1, lk1, lq2, lk2, lam);

  pack_A_f32<<<1024, 256, 0, stream>>>(x,  ApX, 128, 64, DMODEL);
  pack_B_f32<<<1024, 256, 0, stream>>>(Wq, BpQ, 64, 128, DMODEL);
  pack_B_f32<<<1024, 256, 0, stream>>>(Wk, BpK, 64, 128, DMODEL);
  pack_B_f32<<<1024, 256, 0, stream>>>(Wv, BpV, 64, 128, DMODEL);
  pack_B_f32<<<1024, 256, 0, stream>>>(Wo, BpO, 64, 128, DMODEL);

  dim3 ggrid(DMODEL / 128, S_SEQ / 128);  // (16, 16)
  gemm_wmma<_Float16><<<ggrid, 256, 0, stream>>>(ApX, BpQ, Qb, S_SEQ, DMODEL, DMODEL);
  gemm_wmma<_Float16><<<ggrid, 256, 0, stream>>>(ApX, BpK, Kb, S_SEQ, DMODEL, DMODEL);
  gemm_wmma<_Float16><<<ggrid, 256, 0, stream>>>(ApX, BpV, Vb, S_SEQ, DMODEL, DMODEL);

  pack_Khead<<<1024, 256, 0, stream>>>(Kb, Kp);
  pack_Vhead<<<1024, 256, 0, stream>>>(Vb, Vp);

  dim3 agrid(S_SEQ / 16, NHEAD);  // (128, 16)
  diff_attn_kernel<<<agrid, 32, 0, stream>>>(Qb, Kp, Vp, lam, gnw, gnb, ApAtt);

  gemm_wmma<float><<<ggrid, 256, 0, stream>>>(ApAtt, BpO, out, S_SEQ, DMODEL, DMODEL);
}
